// R3DAttention_26070451486878
// MI455X (gfx1250) — matrix operand order repaired and checked by reading, not compiled, not run on hardware
//
#include <hip/hip_runtime.h>

// ---------------------------------------------------------------------------
// Problem constants (from reference): H=1024, NH=16, HS=64, B=8, P=1024
// ---------------------------------------------------------------------------
#define HDIM  1024
#define NHEAD 16
#define HSZ   64
#define BATCH 8
#define SEQ   1024
#define MROWS (BATCH * SEQ)   // 8192

typedef __attribute__((ext_vector_type(16))) __bf16 bf16x16;
typedef __attribute__((ext_vector_type(8)))  float  floatx8;
typedef __attribute__((ext_vector_type(4)))  unsigned int uintx4;   // 16B chunk

union BF16Frag { bf16x16 v; unsigned short u[16]; uintx4 q[2]; };
union U16x8    { uintx4 v; unsigned short u[8]; };

__device__ __forceinline__ unsigned short f2bf(float f) {
    unsigned int u = __float_as_uint(f);
    u += 0x7FFFu + ((u >> 16) & 1u);          // round to nearest even
    return (unsigned short)(u >> 16);
}

// Async global->LDS copy of 16 bytes per lane (CDNA5 GLOBAL_LOAD_ASYNC_TO_LDS,
// tracked by ASYNCcnt). LDS offset = low 32 bits of the generic pointer
// (ISA aperture rule).
__device__ __forceinline__ void async_copy_b128(const void* gptr, void* lptr) {
    const unsigned long long g = (unsigned long long)(uintptr_t)gptr;
    const unsigned int       l = (unsigned int)(uintptr_t)lptr;
    asm volatile("global_load_async_to_lds_b128 %0, %1, off"
                 :: "v"(l), "v"(g) : "memory");
}
__device__ __forceinline__ void async_wait0() {
    asm volatile("s_wait_asynccnt 0x0" ::: "memory");
}

// ---------------------------------------------------------------------------
// GEMM + bias: C[M,N] = A[M,K] * B[K,N] + bias[N]
// Block: 128 thr (4 waves), tile 64(M) x 64(N), K-step 32.
// Wave grid 2x2; each wave owns a 32x32 macro-tile = 2x2 WMMA accumulators,
// so 4 v_wmma per K-step with A/B fragment reuse.
// B is stored TRANSPOSED in LDS ([n][k]) so B fragments are contiguous b128.
// ---------------------------------------------------------------------------
template <bool A_BF16, bool OUT_BF16>
__global__ __launch_bounds__(128) void gemm_bias_wmma(
    const void* __restrict__ Ap, const float* __restrict__ Bp,
    const float* __restrict__ bias, void* __restrict__ Cp,
    int M, int N, int K)
{
    __shared__ unsigned short As[64][32];    // [m][k]
    __shared__ unsigned short BsT[64][32];   // [n][k]  (transposed)

    const int tid  = threadIdx.x;
    const int lane = tid & 31;
    const int wave = tid >> 5;         // 0..3
    const int wy   = wave >> 1;        // 32-row group
    const int wx   = wave & 1;         // 32-col group
    const int hf   = lane >> 4;
    const int l16  = lane & 15;
    const int m0   = blockIdx.y * 64;
    const int n0   = blockIdx.x * 64;

    const int a_row = tid >> 1;            // 0..63
    const int a_kk  = (tid & 1) * 16;      // 16 elems / thread
    const int b_k   = tid >> 2;            // 0..31
    const int b_nn  = (tid & 3) * 16;      // 16 elems / thread

    floatx8 acc[2][2];
    #pragma unroll
    for (int i = 0; i < 2; ++i)
        #pragma unroll
        for (int j = 0; j < 2; ++j)
            acc[i][j] = (floatx8){0.f,0.f,0.f,0.f,0.f,0.f,0.f,0.f};

    for (int k0 = 0; k0 < K; k0 += 32) {
        // ---- stage A tile (64x32) ----
        if constexpr (A_BF16) {
            const unsigned short* A = (const unsigned short*)Ap;
            const uintx4* src = (const uintx4*)&A[(size_t)(m0 + a_row) * K + k0 + a_kk];
            *(uintx4*)&As[a_row][a_kk]     = src[0];
            *(uintx4*)&As[a_row][a_kk + 8] = src[1];
        } else {
            const float* A = (const float*)Ap + (size_t)(m0 + a_row) * K + k0 + a_kk;
            unsigned int* dst = (unsigned int*)&As[a_row][a_kk];
            #pragma unroll
            for (int j = 0; j < 8; ++j)
                dst[j] = (unsigned)f2bf(A[2 * j]) | ((unsigned)f2bf(A[2 * j + 1]) << 16);
        }
        // ---- stage B tile (32x64) transposed into BsT[n][k] ----
        {
            const float* Bg = Bp + (size_t)(k0 + b_k) * N + n0 + b_nn;
            #pragma unroll
            for (int j = 0; j < 16; ++j)
                BsT[b_nn + j][b_k] = f2bf(Bg[j]);
        }
        __syncthreads();

        // ---- fragments: all contiguous b128 LDS loads ----
        BF16Frag af[2], bfr[2];
        #pragma unroll
        for (int s = 0; s < 2; ++s) {
            const int ar = wy * 32 + s * 16 + l16;
            af[s].q[0] = *(const uintx4*)&As[ar][hf * 8];       // K = hf*8..+7
            af[s].q[1] = *(const uintx4*)&As[ar][16 + hf * 8];  // K = 16+hf*8..+7
            const int bc = wx * 32 + s * 16 + l16;
            bfr[s].q[0] = *(const uintx4*)&BsT[bc][hf * 8];       // K = hf*8..+7
            bfr[s].q[1] = *(const uintx4*)&BsT[bc][16 + hf * 8];  // K = 16+hf*8..+7
        }
        #pragma unroll
        for (int sm = 0; sm < 2; ++sm)
            #pragma unroll
            for (int sn = 0; sn < 2; ++sn)
                acc[sm][sn] = __builtin_amdgcn_wmma_f32_16x16x32_bf16(
                    false, af[sm].v, false, bfr[sn].v, (short)0, acc[sm][sn], false, false);
        __syncthreads();
    }

    #pragma unroll
    for (int sm = 0; sm < 2; ++sm)
        #pragma unroll
        for (int sn = 0; sn < 2; ++sn)
            #pragma unroll
            for (int r = 0; r < 8; ++r) {
                const int m = m0 + wy * 32 + sm * 16 + r + hf * 8;
                const int n = n0 + wx * 32 + sn * 16 + l16;
                const float val = acc[sm][sn][r] + bias[n];
                if constexpr (OUT_BF16)
                    ((unsigned short*)Cp)[(size_t)m * N + n] = f2bf(val);
                else
                    ((float*)Cp)[(size_t)m * N + n] = val;
            }
}

// ---------------------------------------------------------------------------
// Flash-style attention. One block = (b, head, 32 query rows), 128 thr = 4 waves
// 2x2: wy -> 16-row half, wx -> 16-key / 32-hs half.
// Q,K tiles arrive via GLOBAL_LOAD_ASYNC_TO_LDS (bf16, no conversion needed);
// V is transposed into LDS ([hs][key]) so attn@V B-fragments are b128 loads.
// ---------------------------------------------------------------------------
__global__ __launch_bounds__(128) void attn_wmma(
    const unsigned short* __restrict__ qkv,   // [B*P][3*H] bf16
    const float* __restrict__ mask,           // [P][P]
    unsigned short* __restrict__ z)           // [B*P][H] bf16
{
    __shared__ unsigned short Qt[32][HSZ];    // [qrow][hs]
    __shared__ unsigned short Kt[32][HSZ];    // [key][hs]
    __shared__ unsigned short VtT[HSZ][32];   // [hs][key] (transposed)
    __shared__ float          Sst[32][32];
    __shared__ unsigned short Pt[32][32];
    __shared__ float rowmax[32], rowsum[32], rowscale[32];

    const int tid  = threadIdx.x;
    const int lane = tid & 31;
    const int wave = tid >> 5;
    const int wy   = wave >> 1;
    const int wx   = wave & 1;
    const int hf   = lane >> 4;
    const int l16  = lane & 15;

    const int bid   = blockIdx.x;
    const int qt    = bid & 31;
    const int nh    = (bid >> 5) & (NHEAD - 1);
    const int b     = bid >> 9;
    const int qbase = qt * 32;
    const int LDQ   = 3 * HDIM;
    const size_t rowb = (size_t)b * SEQ;

    // ---- Q tile: async global->LDS, 16B per lane, 256 chunks ----
    #pragma unroll
    for (int it = 0; it < 2; ++it) {
        const int c = tid + it * 128;          // 0..255
        const int r = c >> 3, c8 = (c & 7) * 8;
        async_copy_b128(&qkv[(rowb + qbase + r) * LDQ + nh * HSZ + c8], &Qt[r][c8]);
    }
    if (tid < 32) { rowmax[tid] = -__builtin_inff(); rowsum[tid] = 0.f; }
    async_wait0();
    __syncthreads();

    // Q A-fragments for the two 32-wide K-steps over HS=64 (kept in VGPRs).
    BF16Frag aq[2];
    #pragma unroll
    for (int ks = 0; ks < 2; ++ks) {
        const int qr = wy * 16 + l16;
        aq[ks].q[0] = *(const uintx4*)&Qt[qr][ks * 32 + hf * 8];
        aq[ks].q[1] = *(const uintx4*)&Qt[qr][ks * 32 + 16 + hf * 8];
    }

    floatx8 o0 = {0.f,0.f,0.f,0.f,0.f,0.f,0.f,0.f};
    floatx8 o1 = {0.f,0.f,0.f,0.f,0.f,0.f,0.f,0.f};

    for (int kt = 0; kt < SEQ / 32; ++kt) {
        const int kb = kt * 32;
        // ---- K tile: async global->LDS ----
        #pragma unroll
        for (int it = 0; it < 2; ++it) {
            const int c = tid + it * 128;
            const int r = c >> 3, c8 = (c & 7) * 8;
            async_copy_b128(&qkv[(rowb + kb + r) * LDQ + HDIM + nh * HSZ + c8],
                            &Kt[r][c8]);
        }
        // ---- V tile: vector load + transpose store into VtT[hs][key] ----
        #pragma unroll
        for (int it = 0; it < 2; ++it) {
            const int c = tid + it * 128;
            const int key = c >> 3, h8 = (c & 7) * 8;
            U16x8 t;
            t.v = *(const uintx4*)&qkv[(rowb + kb + key) * LDQ + 2 * HDIM + nh * HSZ + h8];
            #pragma unroll
            for (int j = 0; j < 8; ++j)
                VtT[h8 + j][key] = t.u[j];
        }
        async_wait0();
        __syncthreads();

        // ---- S = Q * K^T (M=qrow, N=key, contraction over hs) ----
        floatx8 s = {0.f,0.f,0.f,0.f,0.f,0.f,0.f,0.f};
        #pragma unroll
        for (int ks = 0; ks < 2; ++ks) {
            BF16Frag bk;
            const int kr = wx * 16 + l16;
            bk.q[0] = *(const uintx4*)&Kt[kr][ks * 32 + hf * 8];
            bk.q[1] = *(const uintx4*)&Kt[kr][ks * 32 + 16 + hf * 8];
            s = __builtin_amdgcn_wmma_f32_16x16x32_bf16(
                false, aq[ks].v, false, bk.v, (short)0, s, false, false);
        }
        #pragma unroll
        for (int r = 0; r < 8; ++r) {
            const int lr = wy * 16 + r + hf * 8;
            const int lc = wx * 16 + l16;
            Sst[lr][lc] = s[r] * 0.125f
                        - mask[(size_t)(qbase + lr) * SEQ + kb + lc];
        }
        __syncthreads();

        // ---- online softmax row update (one thread per row) ----
        if (tid < 32) {
            const float m_old = rowmax[tid];
            float m_new = m_old;
            for (int c = 0; c < 32; ++c) m_new = fmaxf(m_new, Sst[tid][c]);
            const float sc = __expf(m_old - m_new);
            float psum = 0.f;
            for (int c = 0; c < 32; ++c) {
                const float p = __expf(Sst[tid][c] - m_new);
                psum += p;
                Pt[tid][c] = f2bf(p);
            }
            rowsum[tid]   = rowsum[tid] * sc + psum;
            rowmax[tid]   = m_new;
            rowscale[tid] = sc;
        }
        __syncthreads();

        // ---- rescale running O, then O += P * V ----
        #pragma unroll
        for (int r = 0; r < 8; ++r) {
            const float sc = rowscale[wy * 16 + r + hf * 8];
            o0[r] *= sc; o1[r] *= sc;
        }
        BF16Frag ap, bv0, bv1;
        {
            const int pr = wy * 16 + l16;
            ap.q[0] = *(const uintx4*)&Pt[pr][hf * 8];
            ap.q[1] = *(const uintx4*)&Pt[pr][16 + hf * 8];
            const int c0 = wx * 32 + l16;
            bv0.q[0] = *(const uintx4*)&VtT[c0][hf * 8];
            bv0.q[1] = *(const uintx4*)&VtT[c0][16 + hf * 8];
            bv1.q[0] = *(const uintx4*)&VtT[c0 + 16][hf * 8];
            bv1.q[1] = *(const uintx4*)&VtT[c0 + 16][16 + hf * 8];
        }
        o0 = __builtin_amdgcn_wmma_f32_16x16x32_bf16(
            false, ap.v, false, bv0.v, (short)0, o0, false, false);
        o1 = __builtin_amdgcn_wmma_f32_16x16x32_bf16(
            false, ap.v, false, bv1.v, (short)0, o1, false, false);
        __syncthreads();
    }

    #pragma unroll
    for (int r = 0; r < 8; ++r) {
        const int lr  = wy * 16 + r + hf * 8;
        const float inv = 1.0f / rowsum[lr];
        const size_t zrow = (rowb + qbase + lr) * HDIM + nh * HSZ;
        z[zrow + wx * 32 + l16]      = f2bf(o0[r] * inv);
        z[zrow + wx * 32 + 16 + l16] = f2bf(o1[r] * inv);
    }
}

// ---------------------------------------------------------------------------
// Launch: qkv proj -> flash attention -> output proj
// Workspace: qkv bf16 [8192][3072] (48 MiB) + z bf16 [8192][1024] (16 MiB)
// ---------------------------------------------------------------------------
extern "C" void kernel_launch(void* const* d_in, const int* in_sizes, int n_in,
                              void* d_out, int out_size, void* d_ws, size_t ws_size,
                              hipStream_t stream) {
    (void)in_sizes; (void)n_in; (void)out_size; (void)ws_size;
    const float* x    = (const float*)d_in[0];
    const float* mask = (const float*)d_in[1];
    const float* Wqkv = (const float*)d_in[2];
    const float* bqkv = (const float*)d_in[3];
    const float* Wo   = (const float*)d_in[4];
    const float* bo   = (const float*)d_in[5];
    float* out = (float*)d_out;

    unsigned short* qkv = (unsigned short*)d_ws;                 // 8192*3072 bf16
    unsigned short* z   = qkv + (size_t)MROWS * (3 * HDIM);      // 8192*1024 bf16

    // 1) qkv = x @ Wqkv + bqkv  (fp32 in, bf16 out)
    gemm_bias_wmma<false, true>
        <<<dim3((3 * HDIM) / 64, MROWS / 64), dim3(128), 0, stream>>>(
            x, Wqkv, bqkv, qkv, MROWS, 3 * HDIM, HDIM);

    // 2) flash attention per (b, head, 32-row q tile)
    attn_wmma<<<dim3(BATCH * NHEAD * (SEQ / 32)), dim3(128), 0, stream>>>(
        qkv, mask, z);

    // 3) out = z @ Wo + bo  (bf16 in, fp32 out)
    gemm_bias_wmma<true, false>
        <<<dim3(HDIM / 64, MROWS / 64), dim3(128), 0, stream>>>(
            z, Wo, bo, out, MROWS, HDIM, HDIM);
}
